// GATBlock_16698832847420
// MI455X (gfx1250) — compile-verified, matrix operand order checked by reading, not run on hardware
//
#include <hip/hip_runtime.h>
#include <hip/hip_bf16.h>

typedef __attribute__((ext_vector_type(16))) _Float16 v16h;
typedef __attribute__((ext_vector_type(8)))  float    v8f;

#define NEG_SLOPE 0.2f

// ---- monotone float <-> uint encoding for atomicMax on floats ----
__device__ __forceinline__ unsigned enc_f(float f) {
    unsigned u = __float_as_uint(f);
    return (u & 0x80000000u) ? ~u : (u | 0x80000000u);
}
__device__ __forceinline__ float dec_f(unsigned u) {
    return (u & 0x80000000u) ? __uint_as_float(u ^ 0x80000000u)
                             : __uint_as_float(~u);
}

// ---- f32 -> f16 cast (optionally with ReLU fused) ----
__global__ void k_cast(const float* __restrict__ in, _Float16* __restrict__ out,
                       int n, int relu) {
    int tid = blockIdx.x * blockDim.x + threadIdx.x;
    if (tid >= n) return;
    float v = in[tid];
    if (relu) v = v > 0.f ? v : 0.f;
    out[tid] = (_Float16)v;
}

// ---- WMMA GEMM: HP[N,128] = A[N,K] (f16) * W[128,K]^T (f32 -> f16 in LDS) ----
// 256 threads = 8 waves; block tile = 128 rows x 128 cols; wave tile = 16 x 128.
__global__ __launch_bounds__(256) void k_gemm_wmma(
    const _Float16* __restrict__ A,   // [Npad, K] row-major f16
    const float*    __restrict__ W,   // [128, K] row-major f32 (row j = column j of B)
    float*          __restrict__ HP,  // [Npad, 128] f32
    int K, int nRows)
{
    __shared__ _Float16 sW[128 * 128];   // 32 KB max (K<=128)

    // stage W -> LDS as f16 (row-major [128,K]); broadcast-reused by all 8 waves
    for (int i = threadIdx.x; i < 128 * K; i += 256)
        sW[i] = (_Float16)W[i];
    __syncthreads();

    const int wave = threadIdx.x >> 5;
    const int lane = threadIdx.x & 31;
    const int m    = lane & 15;        // M index (A) / N index (B,D)
    const int hi   = lane >> 4;        // half-wave select
    const int rowBase = blockIdx.x * 128 + wave * 16;

    v8f acc[8] = {};

    const unsigned int* Arow =
        (const unsigned int*)(A + (size_t)(rowBase + m) * K);

    for (int k0 = 0; k0 < K; k0 += 32) {
        union { v16h v; unsigned int u[8]; } afrag;
#pragma unroll
        for (int q = 0; q < 8; ++q) {
            // 16-bit A 16x32 layout: VGPR q holds K pair base:
            // q<4 -> 2q ; q>=4 -> 16+2(q-4) ; +8 for lanes 16..31
            int kb = ((q & 3) * 2) + ((q >> 2) * 16) + hi * 8;
            afrag.u[q] = Arow[(k0 + kb) >> 1];
        }
#pragma unroll
        for (int t = 0; t < 8; ++t) {
            union { v16h v; unsigned int u[8]; } bfrag;
            const unsigned int* Brow =
                (const unsigned int*)(sW + (size_t)(t * 16 + m) * K);
#pragma unroll
            for (int q = 0; q < 8; ++q) {
                int kb = ((q & 3) * 2) + ((q >> 2) * 16) + hi * 8;
                bfrag.u[q] = Brow[(k0 + kb) >> 1];
            }
            acc[t] = __builtin_amdgcn_wmma_f32_16x16x32_f16(
                false, afrag.v, false, bfrag.v, (short)0, acc[t], false, false);
        }
    }

    // D layout: VGPR r -> row r + 8*hi, col = 16*t + m
#pragma unroll
    for (int t = 0; t < 8; ++t) {
#pragma unroll
        for (int r = 0; r < 8; ++r) {
            int row = rowBase + r + 8 * hi;
            if (row < nRows)
                HP[(size_t)row * 128 + t * 16 + m] = acc[t][r];
        }
    }
}

// ---- attention logits: ALS/ALD[n,h] = sum_c HP[n,h,c]*a_{src,dst}[h,c] ----
// one block per node, 4 waves = 4 heads, 32-lane shuffle reduction
__global__ void k_attn(const float* __restrict__ HP,
                       const float* __restrict__ a_src,
                       const float* __restrict__ a_dst,
                       float* __restrict__ ALS, float* __restrict__ ALD) {
    int n = blockIdx.x;
    int h = threadIdx.x >> 5, lane = threadIdx.x & 31;
    float v  = HP[(size_t)n * 128 + h * 32 + lane];
    float vs = v * a_src[h * 32 + lane];
    float vd = v * a_dst[h * 32 + lane];
#pragma unroll
    for (int off = 16; off; off >>= 1) {
        vs += __shfl_down(vs, off, 32);
        vd += __shfl_down(vd, off, 32);
    }
    if (lane == 0) { ALS[n * 4 + h] = vs; ALD[n * 4 + h] = vd; }
}

// ---- per-layer node-state init: OUT = bias broadcast, MENC = enc(-inf floor), S = 0
__global__ void k_init_node(float* __restrict__ OUT, const float* __restrict__ bias,
                            unsigned* __restrict__ MENC, float* __restrict__ S, int N) {
    int tid = blockIdx.x * blockDim.x + threadIdx.x;
    if (tid >= N * 128) return;
    int n = tid >> 7, c = tid & 127;
    OUT[tid] = bias[c];
    if (c < 4) { MENC[n * 4 + c] = 0u; S[n * 4 + c] = 0.f; }
}

// ---- pass 1: leaky-relu score, store, segment-max over dst ----
__global__ void k_edge_score(const int* __restrict__ ei, int E, int Etot,
                             const float* __restrict__ ALS, const float* __restrict__ ALD,
                             float* __restrict__ P, unsigned* __restrict__ MENC) {
    int tid = blockIdx.x * blockDim.x + threadIdx.x;
    if (tid >= Etot * 4) return;
    int e = tid >> 2, h = tid & 3;
    int s = (e < E) ? ei[e]     : (e - E);
    int d = (e < E) ? ei[E + e] : (e - E);
    float x = ALS[s * 4 + h] + ALD[d * 4 + h];
    x = x > 0.f ? x : NEG_SLOPE * x;
    P[tid] = x;
    atomicMax(&MENC[d * 4 + h], enc_f(x));   // GLOBAL_ATOMIC_MAX_U32
}

// ---- pass 2: p = exp(score - max), segment-sum over dst ----
__global__ void k_edge_exp(const int* __restrict__ ei, int E, int Etot,
                           float* __restrict__ P, const unsigned* __restrict__ MENC,
                           float* __restrict__ S) {
    int tid = blockIdx.x * blockDim.x + threadIdx.x;
    if (tid >= Etot * 4) return;
    int e = tid >> 2, h = tid & 3;
    int d = (e < E) ? ei[E + e] : (e - E);
    float p = __expf(P[tid] - dec_f(MENC[d * 4 + h]));
    P[tid] = p;
    atomicAdd(&S[d * 4 + h], p);             // GLOBAL_ATOMIC_ADD_F32
}

// ---- pass 3: OUT[dst] += (p/s) * HP[src]; 128 threads per edge ----
__global__ void k_edge_aggr(const int* __restrict__ ei, int E, int Etot,
                            const float* __restrict__ P, const float* __restrict__ S,
                            const float* __restrict__ HP, float* __restrict__ OUT) {
    int tid = blockIdx.x * blockDim.x + threadIdx.x;
    int e = tid >> 7;
    if (e >= Etot) return;
    int c = tid & 127, h = c >> 5;
    int s = (e < E) ? ei[e]     : (e - E);
    int d = (e < E) ? ei[E + e] : (e - E);
    float alpha = P[e * 4 + h] / (S[d * 4 + h] + 1e-16f);
    atomicAdd(&OUT[(size_t)d * 128 + c], alpha * HP[(size_t)s * 128 + c]);
}

// ---- global mean pool ----
__global__ void k_pool_init(float* __restrict__ POOL, float* __restrict__ CNT, int G) {
    int tid = blockIdx.x * blockDim.x + threadIdx.x;
    if (tid < G * 128) POOL[tid] = 0.f;
    if (tid < G) CNT[tid] = 0.f;
}
__global__ void k_pool_sum(const float* __restrict__ OUT, const int* __restrict__ batch,
                           float* __restrict__ POOL, float* __restrict__ CNT, int N) {
    int tid = blockIdx.x * blockDim.x + threadIdx.x;
    if (tid >= N * 128) return;
    int n = tid >> 7, c = tid & 127;
    int g = batch[n];
    atomicAdd(&POOL[g * 128 + c], OUT[tid]);
    if (c == 0) atomicAdd(&CNT[g], 1.f);
}
__global__ void k_pool_final(const float* __restrict__ POOL, const float* __restrict__ CNT,
                             float* __restrict__ out, int G) {
    int tid = blockIdx.x * blockDim.x + threadIdx.x;
    if (tid >= G * 128) return;
    out[tid] = POOL[tid] / fmaxf(CNT[tid >> 7], 1.f);
}

static inline int cdiv(long long a, long long b) { return (int)((a + b - 1) / b); }

extern "C" void kernel_launch(void* const* d_in, const int* in_sizes, int n_in,
                              void* d_out, int out_size, void* d_ws, size_t ws_size,
                              hipStream_t stream) {
    const float* x     = (const float*)d_in[0];
    const int*   ei    = (const int*)d_in[1];
    const int*   batch = (const int*)d_in[3];
    const float* Wl[3] = {(const float*)d_in[4], (const float*)d_in[8],  (const float*)d_in[12]};
    const float* AS[3] = {(const float*)d_in[5], (const float*)d_in[9],  (const float*)d_in[13]};
    const float* AD[3] = {(const float*)d_in[6], (const float*)d_in[10], (const float*)d_in[14]};
    const float* BI[3] = {(const float*)d_in[7], (const float*)d_in[11], (const float*)d_in[15]};

    const int N    = in_sizes[0] / 64;
    const int E    = in_sizes[1] / 2;
    const int Etot = E + N;
    const int G    = out_size / 128;
    const int Npad = cdiv(N, 128) * 128;

    // ---- carve workspace ----
    size_t off = 0;
    auto carve = [&](size_t bytes) -> void* {
        void* p = (char*)d_ws + off;
        off += (bytes + 255) & ~(size_t)255;
        return p;
    };
    _Float16* A16  = (_Float16*)carve((size_t)Npad * 128 * sizeof(_Float16)); // 25.6 MB
    float*    HP   = (float*)   carve((size_t)Npad * 128 * sizeof(float));    // 51.2 MB
    float*    OUT  = (float*)   carve((size_t)N    * 128 * sizeof(float));    // 51.2 MB
    float*    ALS  = (float*)   carve((size_t)N * 4 * sizeof(float));
    float*    ALD  = (float*)   carve((size_t)N * 4 * sizeof(float));
    unsigned* MENC = (unsigned*)carve((size_t)N * 4 * sizeof(unsigned));
    float*    S    = (float*)   carve((size_t)N * 4 * sizeof(float));
    float*    P    = (float*)   carve((size_t)Etot * 4 * sizeof(float));      // 14.4 MB
    float*    POOL = (float*)   carve((size_t)G * 128 * sizeof(float));
    float*    CNT  = (float*)   carve((size_t)G * sizeof(float));
    (void)ws_size; (void)n_in;

    const int Kdim[3] = {64, 128, 128};
    const int TPB = 256;

    // layer-0 input cast: x [N,64] f32 -> f16
    k_cast<<<cdiv((long long)N * 64, TPB), TPB, 0, stream>>>(x, A16, N * 64, 0);

    for (int l = 0; l < 3; ++l) {
        const int K = Kdim[l];
        k_gemm_wmma<<<cdiv(Npad, 128), TPB, 0, stream>>>(A16, Wl[l], HP, K, N);
        k_attn<<<N, 128, 0, stream>>>(HP, AS[l], AD[l], ALS, ALD);
        k_init_node<<<cdiv((long long)N * 128, TPB), TPB, 0, stream>>>(OUT, BI[l], MENC, S, N);
        k_edge_score<<<cdiv((long long)Etot * 4, TPB), TPB, 0, stream>>>(ei, E, Etot, ALS, ALD, P, MENC);
        k_edge_exp<<<cdiv((long long)Etot * 4, TPB), TPB, 0, stream>>>(ei, E, Etot, P, MENC, S);
        k_edge_aggr<<<cdiv((long long)Etot * 128, TPB), TPB, 0, stream>>>(ei, E, Etot, P, S, HP, OUT);
        if (l < 2) // ReLU + recast for next layer's WMMA A operand
            k_cast<<<cdiv((long long)N * 128, TPB), TPB, 0, stream>>>(OUT, A16, N * 128, 1);
    }

    k_pool_init<<<cdiv((long long)G * 128, TPB), TPB, 0, stream>>>(POOL, CNT, G);
    k_pool_sum<<<cdiv((long long)N * 128, TPB), TPB, 0, stream>>>(OUT, batch, POOL, CNT, N);
    k_pool_final<<<cdiv((long long)G * 128, TPB), TPB, 0, stream>>>(POOL, CNT, (float*)d_out, G);
}
